// BorderHead_12455405158549
// MI455X (gfx1250) — matrix operand order repaired and checked by reading, not compiled, not run on hardware
//
#include <hip/hip_runtime.h>
#include <hip/hip_bf16.h>

// BorderHead on MI455X (gfx1250): implicit-GEMM conv via bf16 WMMA.
// conv1 is M=100352, N=256, K=20736 -> ~1.07 TFLOP (compute-bound; total HBM
// traffic < 1 GB ~ 40us at 23.3 TB/s vs ~200us of bf16 WMMA) -> matrix cores.
// Data movement: B tiles via gfx1250 async global->LDS DMA (ASYNCcnt),
// A tiles synthesized on the fly (feats = x - shift(x)) with global loads
// issued one K-step ahead into registers; triple-buffered LDS, one
// LDS-scoped split barrier per K-step (no loadcnt drain -> prefetch loads
// stay in flight across the WMMA phase).

typedef __bf16 bf16_t;
typedef __attribute__((ext_vector_type(16))) __bf16 v16bf;
typedef __attribute__((ext_vector_type(8)))  __bf16 v8bf;
typedef __attribute__((ext_vector_type(8)))  float  v8f;

#define NIMG 32
#define C_IN 256
#define HH 56
#define WW 56
#define HWSZ (HH * WW)          // 3136
#define KFEAT (9 * C_IN)        // 2304
#define COUT 256
#define NCLS 80
#define MTOT (NIMG * HWSZ)      // 100352 = 784 * 128
#define KSTEPS 648              // 9 taps * 9 groups * 8 chan-chunks (K=32 each)

#define LDA 40                  // padded LDS row stride (conflict-free b128 frag reads)
#define LDW 264                 // padded Ws row stride in predictor

// (di+1), (dj+1) for groups 0..8 packed 2 bits each (g uniform -> SALU only).
// groups: 0 = identity, 1..8 = shifts (1,0)(-1,0)(0,1)(0,-1)(1,1)(1,-1)(-1,1)(-1,-1)
#define DIP 0x2949u
#define DJP 0x8895u

static __device__ __forceinline__ v8f zero_v8f() {
  v8f z;
#pragma unroll
  for (int i = 0; i < 8; ++i) z[i] = 0.0f;
  return z;
}
static __device__ __forceinline__ v8bf zero_v8bf() {
  v8bf z;
#pragma unroll
  for (int i = 0; i < 8; ++i) z[i] = (__bf16)0.0f;
  return z;
}
static __device__ __forceinline__ v16bf cat8(v8bf lo, v8bf hi) {
  return __builtin_shufflevector(lo, hi, 0, 1, 2, 3, 4, 5, 6, 7,
                                 8, 9, 10, 11, 12, 13, 14, 15);
}

// gfx1250 async DMA: 16B global -> LDS, tracked by ASYNCcnt (not LOADcnt).
// VDST operand carries the LDS byte address (flat-shared addr[31:0] == LDS
// offset per the aperture rules), VADDR is the 64-bit global address.
static __device__ __forceinline__ void async_copy16(__bf16* lds_dst,
                                                    const __bf16* gsrc) {
  unsigned ldsoff = (unsigned)(unsigned long long)(uintptr_t)lds_dst;
  asm volatile("global_load_async_to_lds_b128 %0, %1, off"
               :
               : "v"(ldsoff), "v"(gsrc)
               : "memory");
}

// Workgroup split-barrier that drains ONLY DScnt (LDS publication). Unlike
// __syncthreads() it does NOT wait LOADcnt, so per-thread register prefetch
// loads issued before the barrier stay in flight across the WMMA phase.
// (ASYNCcnt for the DMA tile is waited explicitly by the caller.)
static __device__ __forceinline__ void wg_barrier_lds() {
  asm volatile(
      "s_wait_dscnt 0x0\n\t"
      "s_barrier_signal -1\n\t"
      "s_barrier_wait -1" ::: "memory");
}

// ---------------------------------------------------------------------------
// Kernel 1: x NCHW f32 -> xT NHWC bf16, LDS-transposed so both sides coalesce.
// ---------------------------------------------------------------------------
__global__ __launch_bounds__(256) void bh_pack_x(const float* __restrict__ x,
                                                 __bf16* __restrict__ xT) {
  __shared__ float tile[64][65];
  const int hwb = blockIdx.x * 64;
  const int cb = blockIdx.y * 64;
  const int n = blockIdx.z;
  const int t = threadIdx.x;
  const float* xs = x + (size_t)n * C_IN * HWSZ;
#pragma unroll
  for (int i = 0; i < 16; ++i) {
    int idx = t + i * 256;
    int cy = idx >> 6, hwx = idx & 63;
    tile[cy][hwx] = xs[(size_t)(cb + cy) * HWSZ + hwb + hwx];
  }
  __syncthreads();
  __bf16* xd = xT + (size_t)n * HWSZ * C_IN;
#pragma unroll
  for (int i = 0; i < 16; ++i) {
    int idx = t + i * 256;
    int hwy = idx >> 6, cx = idx & 63;
    xd[(size_t)(hwb + hwy) * C_IN + cb + cx] = (__bf16)tile[cx][hwy];
  }
}

// ---------------------------------------------------------------------------
// Kernel 2: w1 [Cout, 2304, 3, 3] f32 -> Bpack [tap][Cout][2304] bf16.
// ---------------------------------------------------------------------------
__global__ __launch_bounds__(256) void bh_pack_w1(const float* __restrict__ w1,
                                                  __bf16* __restrict__ Bpack) {
  int idx = blockIdx.x * 256 + threadIdx.x;  // 9*256*2304 = 5,308,416
  if (idx >= 9 * COUT * KFEAT) return;
  int ic = idx % KFEAT;
  int rest = idx / KFEAT;
  int oc = rest % COUT;
  int tap = rest / COUT;
  Bpack[idx] = (__bf16)w1[(size_t)(oc * KFEAT + ic) * 9 + tap];
}

// ---------------------------------------------------------------------------
// Kernel 3: w2 [80,256] f32 -> bf16.
// ---------------------------------------------------------------------------
__global__ __launch_bounds__(256) void bh_pack_w2(const float* __restrict__ w2,
                                                  __bf16* __restrict__ W2p) {
  int idx = blockIdx.x * 256 + threadIdx.x;  // 80*256 = 20480
  if (idx < NCLS * COUT) W2p[idx] = (__bf16)w2[idx];
}

// ---------------------------------------------------------------------------
// Kernel 4: conv1 implicit GEMM. Block tile 128M x 128N, 8 waves (4M x 2N),
// wave tile 32x64 = 2x4 v_wmma_f32_16x16x32_bf16. Flat 648-step K loop.
// Pipeline per iter k (one LDS-scoped barrier per step, triple-buffered LDS):
//   consume A regs (issued at iter k-1; s_wait_loadcnt lands here, a full
//     WMMA phase after issue) -> diff -> ds_store As[k%3]
//   issue  A global loads for step k+1 into regs
//   issue  async B DMA for step k+1 -> Bs[(k+1)%3]
//   s_wait_asynccnt 2 (batch k landed) ; dscnt-only barrier ; WMMA buf k%3
// ---------------------------------------------------------------------------
struct APend {
  v8bf a0, a1, c0, c1;
  float scale;
};

__global__ __launch_bounds__(256) void bh_conv1_wmma(
    const __bf16* __restrict__ xT, const __bf16* __restrict__ Bpack,
    const float* __restrict__ b1, __bf16* __restrict__ h1T) {
  __shared__ __bf16 As[3][128 * LDA];
  __shared__ __bf16 Bs[3][128 * LDA];

  const int t = threadIdx.x;
  const int lane = t & 31;
  const int wv = t >> 5;  // 0..7
  const int wm = wv >> 1; // 0..3 (M)
  const int wn = wv & 1;  // 0..1 (N)
  const int Mbase = blockIdx.x * 128;
  const int Nbase = blockIdx.y * 128;

  // Per-thread staging coords: thread covers (row = t/2, 16-chan half = t&1)
  const int arow = t >> 1;
  const int ahalf = t & 1;
  const int m = Mbase + arow;
  const int nimg = m / HWSZ;
  const int hw = m % HWSZ;
  const int hy = hw / WW;
  const int wx = hw % WW;
  const __bf16* xTn = xT + (size_t)nimg * HWSZ * C_IN;
  const __bf16* Brow = Bpack + (size_t)(Nbase + arow) * KFEAT + ahalf * 16;

  v8f acc[2][4];
#pragma unroll
  for (int a = 0; a < 2; ++a)
#pragma unroll
    for (int b = 0; b < 4; ++b) acc[a][b] = zero_v8f();

  // ---- issue step ks: A global loads -> regs, B async DMA -> Bs[buf] ----
  auto issueA = [&](int ks, int buf, APend& P) {
    const int tap = ks / 72;  // uniform (SALU)
    const int rem = ks - tap * 72;
    const int g = rem >> 3;
    const int cc = (rem & 7) << 5;
    const int kh = tap / 3;
    const int kw = tap - kh * 3;
    const int ii = hy + kh - 1;
    const int jj = wx + kw - 1;
    const bool inb = ((unsigned)ii < (unsigned)HH) && ((unsigned)jj < (unsigned)WW);
    const int di = (int)((DIP >> (2 * g)) & 3u) - 1;
    const int dj = (int)((DJP >> (2 * g)) & 3u) - 1;
    const int i2 = ii - di, j2 = jj - dj;
    const bool inb2 =
        inb && ((unsigned)i2 < (unsigned)HH) && ((unsigned)j2 < (unsigned)WW);
    const __bf16* s1 = xTn + (size_t)(ii * WW + jj) * C_IN + ahalf * 16 + cc;
    const __bf16* s2 = xTn + (size_t)(i2 * WW + j2) * C_IN + ahalf * 16 + cc;
    const bool g0 = (g == 0);

    P.a0 = zero_v8bf();
    P.a1 = zero_v8bf();
    P.c0 = zero_v8bf();
    P.c1 = zero_v8bf();
    P.scale = (g >= 5) ? 0.70710678118654752f : 1.0f;
    if (g0 ? inb : inb2) {
      P.a0 = *(const v8bf*)s1;
      P.a1 = *(const v8bf*)(s1 + 8);
      if (!g0) {
        P.c0 = *(const v8bf*)s2;
        P.c1 = *(const v8bf*)(s2 + 8);
      }
    }

    const __bf16* bp = Brow + (size_t)tap * (COUT * KFEAT) + g * C_IN + cc;
    __bf16* bb = &Bs[buf][arow * LDA + ahalf * 16];
    async_copy16(bb, bp);
    async_copy16(bb + 8, bp + 8);
  };

  // ---- consume: branch-free feats diff (c=0, scale=1 on identity group) ----
  auto consumeA = [&](const APend& P, int buf) {
    v8bf r0, r1;
#pragma unroll
    for (int e = 0; e < 8; ++e) {
      r0[e] = (__bf16)(((float)P.a0[e] - (float)P.c0[e]) * P.scale);
      r1[e] = (__bf16)(((float)P.a1[e] - (float)P.c1[e]) * P.scale);
    }
    __bf16* ab = &As[buf][arow * LDA + ahalf * 16];
    *(v8bf*)ab = r0;
    *(v8bf*)(ab + 8) = r1;
  };

  const int mA = lane & 15;
  const int khalf = lane >> 4;

  auto computeTile = [&](int buf) {
    const __bf16* Ac = &As[buf][0];
    const __bf16* Bc = &Bs[buf][0];
    v16bf af[2];
#pragma unroll
    for (int tm = 0; tm < 2; ++tm) {
      const __bf16* p = Ac + (wm * 32 + tm * 16 + mA) * LDA;
      v8bf lo = *(const v8bf*)(p + khalf * 8);
      v8bf hi = *(const v8bf*)(p + 16 + khalf * 8);
      af[tm] = cat8(lo, hi);
    }
#pragma unroll
    for (int tn = 0; tn < 4; ++tn) {
      const __bf16* p = Bc + (wn * 64 + tn * 16 + mA) * LDA;
      v8bf lo = *(const v8bf*)(p + khalf * 8);
      v8bf hi = *(const v8bf*)(p + 16 + khalf * 8);
      v16bf bfr = cat8(lo, hi);
#pragma unroll
      for (int tm = 0; tm < 2; ++tm)
        acc[tm][tn] = __builtin_amdgcn_wmma_f32_16x16x32_bf16(
            false, af[tm], false, bfr, (short)0, acc[tm][tn], false, false);
    }
  };

  int cur = 0;
  auto step = [&](int k, const APend& Pc, APend& Pi) {
    consumeA(Pc, cur);  // ds_store A for step k (loads waited here, issued k-1)
    int nxt = cur + 1;
    if (nxt == 3) nxt = 0;
    if (k + 1 < KSTEPS) {
      issueA(k + 1, nxt, Pi);
      // batch for buf `cur` (2 oldest async instrs) has landed once <=2 remain
      asm volatile("s_wait_asynccnt 0x2" ::: "memory");
    } else {
      asm volatile("s_wait_asynccnt 0x0" ::: "memory");
    }
    wg_barrier_lds();  // dscnt-only: A prefetch loads stay in flight
    computeTile(cur);
    cur = nxt;
  };

  APend P, Q;
  issueA(0, 0, P);
  // 2x unroll with ping-pong pending regs (KSTEPS is even) -> no struct copies
  for (int k = 0; k < KSTEPS; k += 2) {
    step(k, P, Q);
    step(k + 1, Q, P);
  }

  // ---- epilogue: + bias, store bf16 hidden in [M][256] row-major ----
#pragma unroll
  for (int tm = 0; tm < 2; ++tm)
#pragma unroll
    for (int tn = 0; tn < 4; ++tn) {
      const int ng = Nbase + wn * 64 + tn * 16 + (lane & 15);
      const float bias = b1[ng];
#pragma unroll
      for (int r = 0; r < 8; ++r) {
        const int mg = Mbase + wm * 32 + tm * 16 + (lane >> 4) * 8 + r;
        h1T[(size_t)mg * COUT + ng] = (__bf16)(acc[tm][tn][r] + bias);
      }
    }
}

// ---------------------------------------------------------------------------
// Kernel 5: predictor 1x1 conv. M=128/block (8 waves x 16 rows), N=80=5x16,
// K=256 in 8 WMMA steps. w2 resident in LDS; A staged via async DMA.
// ---------------------------------------------------------------------------
__global__ __launch_bounds__(256) void bh_pred_wmma(
    const __bf16* __restrict__ h1T, const __bf16* __restrict__ W2p,
    const float* __restrict__ b2, float* __restrict__ out) {
  __shared__ __bf16 Ws[NCLS * LDW];  // 80 x 264 (padded)
  __shared__ __bf16 As[128 * LDA];

  const int t = threadIdx.x;
  const int lane = t & 31;
  const int wv = t >> 5;  // wave id -> 16-row M sub-tile
  const int Mbase = blockIdx.x * 128;

  // stage W2 into LDS (once per block)
#pragma unroll
  for (int i = 0; i < 10; ++i) {
    int v = t + i * 256;  // vec8 unit id, 2560 total
    int row = v >> 5;     // 32 vec8 per 256-wide row
    int col8 = (v & 31) * 8;
    *(v8bf*)&Ws[row * LDW + col8] = *(const v8bf*)&W2p[row * COUT + col8];
  }

  v8f acc[5];
#pragma unroll
  for (int tn = 0; tn < 5; ++tn) acc[tn] = zero_v8f();

  const int arow = t >> 1, ahalf = t & 1;
  const __bf16* arp = h1T + (size_t)(Mbase + arow) * COUT + ahalf * 16;

  for (int k0 = 0; k0 < COUT; k0 += 32) {
    __syncthreads();  // previous reads done; also orders first Ws use
    __bf16* ab = &As[arow * LDA + ahalf * 16];
    async_copy16(ab, arp + k0);
    async_copy16(ab + 8, arp + k0 + 8);
    asm volatile("s_wait_asynccnt 0x0" ::: "memory");
    __syncthreads();

    const int mA = lane & 15, khalf = lane >> 4;
    const __bf16* pa = &As[(wv * 16 + mA) * LDA];
    v8bf alo = *(const v8bf*)(pa + khalf * 8);
    v8bf ahi = *(const v8bf*)(pa + 16 + khalf * 8);
    v16bf af = cat8(alo, ahi);
#pragma unroll
    for (int tn = 0; tn < 5; ++tn) {
      const __bf16* pb = &Ws[(tn * 16 + mA) * LDW + k0];
      v8bf blo = *(const v8bf*)(pb + khalf * 8);
      v8bf bhi = *(const v8bf*)(pb + 16 + khalf * 8);
      v16bf bfr = cat8(blo, bhi);
      acc[tn] = __builtin_amdgcn_wmma_f32_16x16x32_bf16(
          false, af, false, bfr, (short)0, acc[tn], false, false);
    }
  }

  // ---- epilogue: + bias, scatter to NCHW f32 output ----
#pragma unroll
  for (int tn = 0; tn < 5; ++tn) {
    const int cls = tn * 16 + (lane & 15);
    const float bias = b2[cls];
#pragma unroll
    for (int r = 0; r < 8; ++r) {
      const int mg = Mbase + wv * 16 + (lane >> 4) * 8 + r;
      const int ni = mg / HWSZ;
      const int hw = mg % HWSZ;
      out[((size_t)ni * NCLS + cls) * HWSZ + hw] = acc[tn][r] + bias;
    }
  }
}

// ---------------------------------------------------------------------------
// Launch
// ---------------------------------------------------------------------------
extern "C" void kernel_launch(void* const* d_in, const int* in_sizes, int n_in,
                              void* d_out, int out_size, void* d_ws, size_t ws_size,
                              hipStream_t stream) {
  const float* x = (const float*)d_in[0];
  const float* w1 = (const float*)d_in[1];
  const float* b1 = (const float*)d_in[2];
  const float* w2 = (const float*)d_in[3];
  const float* b2 = (const float*)d_in[4];
  float* out = (float*)d_out;

  const size_t XT_BYTES = (size_t)NIMG * HWSZ * C_IN * 2;  // 51,380,224
  const size_t BP_BYTES = (size_t)9 * COUT * KFEAT * 2;    // 10,616,832
  const size_t W2_BYTES = (size_t)NCLS * COUT * 2;         // 40,960

  char* ws = (char*)d_ws;
  __bf16* xT = (__bf16*)ws;
  __bf16* Bpack = (__bf16*)(ws + XT_BYTES);
  __bf16* W2p = (__bf16*)(ws + XT_BYTES + BP_BYTES);
  __bf16* h1T = (__bf16*)(ws + XT_BYTES + BP_BYTES + W2_BYTES);

  bh_pack_x<<<dim3(HWSZ / 64, C_IN / 64, NIMG), 256, 0, stream>>>(x, xT);
  bh_pack_w1<<<dim3((9 * COUT * KFEAT) / 256), 256, 0, stream>>>(w1, Bpack);
  bh_pack_w2<<<dim3((NCLS * COUT) / 256), 256, 0, stream>>>(w2, W2p);
  bh_conv1_wmma<<<dim3(MTOT / 128, COUT / 128), 256, 0, stream>>>(xT, Bpack, b1, h1T);
  bh_pred_wmma<<<dim3(MTOT / 128), 256, 0, stream>>>(h1T, W2p, b2, out);
}
